// MsCAM_Module_38233798869173
// MI455X (gfx1250) — matrix operand order repaired and checked by reading, not compiled, not run on hardware
//
#include <hip/hip_runtime.h>

#define BDIM 16
#define CDIM 256
#define NDIM 4096
#define LDST  40   // K=32 tiles: 80 B row stride (16B-aligned rows)
#define ELDST 72   // K=64 tiles: 144 B row stride (16B-aligned rows)

typedef __attribute__((ext_vector_type(16))) __bf16 bf16x16;
typedef __attribute__((ext_vector_type(8)))  __bf16 bf16x8;
typedef __attribute__((ext_vector_type(8)))  float  f32x8;
typedef __attribute__((ext_vector_type(4)))  float  f32x4;

__device__ __forceinline__ bf16x8 cvt8(f32x4 a, f32x4 b) {
  bf16x8 r;
  r[0] = (__bf16)a[0]; r[1] = (__bf16)a[1]; r[2] = (__bf16)a[2]; r[3] = (__bf16)a[3];
  r[4] = (__bf16)b[0]; r[5] = (__bf16)b[1]; r[6] = (__bf16)b[2]; r[7] = (__bf16)b[3];
  return r;
}

__device__ __forceinline__ unsigned lds_off(const void* l) {
  return (unsigned)(uintptr_t)l;  // low 32 bits of generic ptr == LDS byte offset
}
// Async copy 16B global -> LDS (per-lane addresses), tracked by ASYNCcnt.
__device__ __forceinline__ void async_b128(const void* g, unsigned loff) {
  asm volatile("global_load_async_to_lds_b128 %0, %1, off"
               :: "v"(loff), "v"(g) : "memory");
}
__device__ __forceinline__ void wait_async0() {
  asm volatile("s_wait_asynccnt 0" ::: "memory");
}

// A fragment (16x32 bf16): lane<16 -> row=lane, K 0..7 & 16..23 ; lane>=16 -> K 8..15 & 24..31
__device__ __forceinline__ bf16x16 frag_a(const __bf16* base, int lane, int stride) {
  const int half = lane >> 4, m = lane & 15;
  const __bf16* p = base + m * stride + half * 8;
  bf16x8 lo = *(const bf16x8*)p;
  bf16x8 hi = *(const bf16x8*)(p + 16);
  return __builtin_shufflevector(lo, hi, 0,1,2,3,4,5,6,7,8,9,10,11,12,13,14,15);
}
// B fragment (32x16 bf16) from an LDS tile stored [N][K]: lane = column, lane<16 -> K 0..7 & 16..23 ; lane>=16 -> K 8..15 & 24..31
__device__ __forceinline__ bf16x16 frag_b(const __bf16* base, int lane, int stride) {
  const int half = lane >> 4, n = lane & 15;
  const __bf16* p = base + n * stride + half * 8;
  bf16x8 lo = *(const bf16x8*)p;
  bf16x8 hi = *(const bf16x8*)(p + 16);
  return __builtin_shufflevector(lo, hi, 0,1,2,3,4,5,6,7,8,9,10,11,12,13,14,15);
}

#define WMMA_BF16(A, Bv, Cv) \
  __builtin_amdgcn_wmma_f32_16x16x32_bf16(false, (A), false, (Bv), (short)0, (Cv), false, false)

// --------------------------------------------------------------------------
// K0: f32 -> bf16 conversion pre-pass (bandwidth-bound, grid-stride)
// --------------------------------------------------------------------------
__global__ __launch_bounds__(256) void k_cvt(const float* __restrict__ src,
                                             __bf16* __restrict__ dst, int n) {
  const int stride = gridDim.x * 256 * 8;
  for (int i = ((int)blockIdx.x * 256 + (int)threadIdx.x) * 8; i < n; i += stride) {
    f32x4 a = *(const f32x4*)(src + i);
    f32x4 b = *(const f32x4*)(src + i + 4);
    *(bf16x8*)(dst + i) = cvt8(a, b);
  }
}

// --------------------------------------------------------------------------
// K1: E[s][b] = q_b (CxN) * k_b^T -> CxC.  bf16 inputs, async-to-LDS staged,
// double-buffered (2x unrolled so buffer index is compile-time), K=64 chunks.
// Block tile 64x128, 8 waves of 32x32.  grid = 3*B*8, block = 256
// --------------------------------------------------------------------------
__global__ __launch_bounds__(256) void k_energy(const __bf16* __restrict__ xh,
                                                const __bf16* __restrict__ yh,
                                                const __bf16* __restrict__ zh,
                                                float* __restrict__ E) {
  __shared__ __bf16 sA[2][64 * ELDST];    // 64 rows x 64 K
  __shared__ __bf16 sB[2][128 * ELDST];   // 128 rows x 64 K
  const int bx = blockIdx.x;
  const int s  = bx / (BDIM * 8);
  const int r0 = bx % (BDIM * 8);
  const int b  = r0 >> 3;
  const int tt = r0 & 7;
  const int c0 = (tt >> 1) * 64;    // E row tile
  const int c1 = (tt & 1) * 128;    // E col tile
  const __bf16* q = xh + (size_t)b * CDIM * NDIM;
  const __bf16* k = (s == 0 ? xh : (s == 1 ? yh : zh)) + (size_t)b * CDIM * NDIM;

  const int t = threadIdx.x, lane = t & 31, wid = t >> 5;
  const int m0 = (wid >> 2) * 32, n0 = (wid & 3) * 32;

  // stagers: each thread copies 2 A rows and 4 B rows (one 16B chunk each)
  const int ar  = t >> 3;            // 0..31
  const int akg = (t & 7) * 8;       // 0..56
  const __bf16* gA0 = q + (size_t)(c0 + ar) * NDIM + akg;
  const __bf16* gA1 = gA0 + (size_t)32 * NDIM;
  const __bf16* gB0 = k + (size_t)(c1 + ar) * NDIM + akg;
  const __bf16* gB1 = gB0 + (size_t)32 * NDIM;
  const __bf16* gB2 = gB0 + (size_t)64 * NDIM;
  const __bf16* gB3 = gB0 + (size_t)96 * NDIM;
  const int ea0 = ar * ELDST + akg, ea1 = (ar + 32) * ELDST + akg;
  const int eb2 = (ar + 64) * ELDST + akg, eb3 = (ar + 96) * ELDST + akg;
  // LDS byte addresses for both buffers, precomputed once
  const unsigned lA0[2] = {lds_off(&sA[0][ea0]), lds_off(&sA[1][ea0])};
  const unsigned lA1[2] = {lds_off(&sA[0][ea1]), lds_off(&sA[1][ea1])};
  const unsigned lB0[2] = {lds_off(&sB[0][ea0]), lds_off(&sB[1][ea0])};
  const unsigned lB1[2] = {lds_off(&sB[0][ea1]), lds_off(&sB[1][ea1])};
  const unsigned lB2[2] = {lds_off(&sB[0][eb2]), lds_off(&sB[1][eb2])};
  const unsigned lB3[2] = {lds_off(&sB[0][eb3]), lds_off(&sB[1][eb3])};

  auto stage = [&](int buf, int ko) {
    async_b128(gA0 + ko, lA0[buf]);  async_b128(gA1 + ko, lA1[buf]);
    async_b128(gB0 + ko, lB0[buf]);  async_b128(gB1 + ko, lB1[buf]);
    async_b128(gB2 + ko, lB2[buf]);  async_b128(gB3 + ko, lB3[buf]);
  };

  f32x8 acc00 = {}, acc01 = {}, acc10 = {}, acc11 = {};
  auto compute = [&](int buf) {
#pragma unroll
    for (int ks = 0; ks < 64; ks += 32) {
      bf16x16 fa0 = frag_a(&sA[buf][m0 * ELDST + ks], lane, ELDST);
      bf16x16 fa1 = frag_a(&sA[buf][(m0 + 16) * ELDST + ks], lane, ELDST);
      bf16x16 fb0 = frag_b(&sB[buf][n0 * ELDST + ks], lane, ELDST);
      bf16x16 fb1 = frag_b(&sB[buf][(n0 + 16) * ELDST + ks], lane, ELDST);
      acc00 = WMMA_BF16(fa0, fb0, acc00);
      acc01 = WMMA_BF16(fa0, fb1, acc01);
      acc10 = WMMA_BF16(fa1, fb0, acc10);
      acc11 = WMMA_BF16(fa1, fb1, acc11);
    }
  };

  stage(0, 0);
  wait_async0();
  __syncthreads();

  for (int kk = 0; kk < NDIM; kk += 128) {   // two 64-K chunks per iteration
    stage(1, kk + 64);                        // kk+64 < NDIM always (kk <= NDIM-128)
    compute(0);
    wait_async0();
    __syncthreads();
    if (kk + 128 < NDIM) stage(0, kk + 128);  // uniform guard, EXEC stays full
    compute(1);
    wait_async0();
    __syncthreads();
  }

  float* Ep = E + (size_t)(s * BDIM + b) * CDIM * CDIM;
  const int half = lane >> 4, nl = lane & 15;
#pragma unroll
  for (int r = 0; r < 8; ++r) {
    const int mA = c0 + m0 + (half ? 8 : 0) + r;
    const int mB = mA + 16;
    const int col = c1 + n0 + nl;
    Ep[(size_t)mA * CDIM + col]      = acc00[r];
    Ep[(size_t)mA * CDIM + col + 16] = acc01[r];
    Ep[(size_t)mB * CDIM + col]      = acc10[r];
    Ep[(size_t)mB * CDIM + col + 16] = acc11[r];
  }
}

// --------------------------------------------------------------------------
// K2: attn = sum_s softmax(rowmax - E_s)  (== softmax(-E_s), min-stabilized)
// Writes attn TRANSPOSED in bf16 so the next GEMM's B-side is K-contiguous.
// grid = B*C rows, block = 256
// --------------------------------------------------------------------------
__global__ __launch_bounds__(256) void k_softmax(const float* __restrict__ E,
                                                 __bf16* __restrict__ attnTh) {
  __shared__ float red[256];
  const int row = blockIdx.x;           // b*C + c
  const int b = row >> 8, c = row & 255;
  const int t = threadIdx.x;            // column d
  float accum = 0.f;
#pragma unroll
  for (int s = 0; s < 3; ++s) {
    const float e = E[((size_t)s * BDIM * CDIM + row) * CDIM + t];
    red[t] = e; __syncthreads();
    for (int off = 128; off > 0; off >>= 1) {
      if (t < off) red[t] = fminf(red[t], red[t + off]);
      __syncthreads();
    }
    const float mn = red[0]; __syncthreads();
    const float v = __expf(mn - e);
    red[t] = v; __syncthreads();
    for (int off = 128; off > 0; off >>= 1) {
      if (t < off) red[t] += red[t + off];
      __syncthreads();
    }
    const float sm = red[0]; __syncthreads();
    accum += v / sm;
  }
  attnTh[((size_t)b * CDIM + t) * CDIM + c] = (__bf16)accum;  // attn^T[d][c]
}

// --------------------------------------------------------------------------
// K3: Mw[b] = conv_w @ attn_b  (both bf16, both async-staged; B from attn^T)
// Block tile 64x128, K=32 chunks fully unrolled.  grid = B*8, block = 256
// --------------------------------------------------------------------------
__global__ __launch_bounds__(256) void k_mix(const __bf16* __restrict__ wh,
                                             const __bf16* __restrict__ attnTh,
                                             __bf16* __restrict__ Mwh) {
  __shared__ __bf16 sA[2][64 * LDST];
  __shared__ __bf16 sB[2][128 * LDST];
  const int bx = blockIdx.x;
  const int b  = bx >> 3;
  const int tt = bx & 7;
  const int o0 = (tt >> 1) * 64;
  const int d0 = (tt & 1) * 128;
  const __bf16* Bt = attnTh + (size_t)b * CDIM * CDIM;

  const int t = threadIdx.x, lane = t & 31, wid = t >> 5;
  const int m0 = (wid >> 2) * 32, n0 = (wid & 3) * 32;

  const int ar = t >> 2, akg = (t & 3) * 8;  // A: 64 rows x 32
  const __bf16* gA  = wh + (size_t)(o0 + ar) * CDIM + akg;
  const __bf16* gB0 = Bt + (size_t)(d0 + ar) * CDIM + akg;   // B: 128 rows x 32
  const __bf16* gB1 = gB0 + (size_t)64 * CDIM;
  const int la = ar * LDST + akg, lb1 = (ar + 64) * LDST + akg;
  const unsigned lAo[2] = {lds_off(&sA[0][la]),  lds_off(&sA[1][la])};
  const unsigned lB0o[2] = {lds_off(&sB[0][la]), lds_off(&sB[1][la])};
  const unsigned lB1o[2] = {lds_off(&sB[0][lb1]), lds_off(&sB[1][lb1])};

  async_b128(gA, lAo[0]);
  async_b128(gB0, lB0o[0]);
  async_b128(gB1, lB1o[0]);
  wait_async0();
  __syncthreads();

  f32x8 acc00 = {}, acc01 = {}, acc10 = {}, acc11 = {};
#pragma unroll
  for (int kk = 0; kk < CDIM; kk += 32) {
    const int cur = (kk >> 5) & 1, nxt = cur ^ 1;
    if (kk + 32 < CDIM) {
      async_b128(gA + kk + 32, lAo[nxt]);
      async_b128(gB0 + kk + 32, lB0o[nxt]);
      async_b128(gB1 + kk + 32, lB1o[nxt]);
    }
    bf16x16 fa0 = frag_a(&sA[cur][m0 * LDST], lane, LDST);
    bf16x16 fa1 = frag_a(&sA[cur][(m0 + 16) * LDST], lane, LDST);
    bf16x16 fb0 = frag_b(&sB[cur][n0 * LDST], lane, LDST);
    bf16x16 fb1 = frag_b(&sB[cur][(n0 + 16) * LDST], lane, LDST);
    acc00 = WMMA_BF16(fa0, fb0, acc00);
    acc01 = WMMA_BF16(fa0, fb1, acc01);
    acc10 = WMMA_BF16(fa1, fb0, acc10);
    acc11 = WMMA_BF16(fa1, fb1, acc11);
    wait_async0();
    __syncthreads();
  }

  __bf16* Mp = Mwh + (size_t)b * CDIM * CDIM;
  const int half = lane >> 4, nl = lane & 15;
#pragma unroll
  for (int r = 0; r < 8; ++r) {
    const int mA = o0 + m0 + (half ? 8 : 0) + r;
    const int mB = mA + 16;
    const int col = d0 + n0 + nl;
    Mp[(size_t)mA * CDIM + col]      = (__bf16)acc00[r];
    Mp[(size_t)mA * CDIM + col + 16] = (__bf16)acc01[r];
    Mp[(size_t)mB * CDIM + col]      = (__bf16)acc10[r];
    Mp[(size_t)mB * CDIM + col + 16] = (__bf16)acc11[r];
  }
}

// --------------------------------------------------------------------------
// K4: out[b,o,n] = gamma*(sum_c Mw[b,o,c]*q[b,c,n] + bias[o]) + x[b,o,n]
// A (Mw, bf16) async-staged; B (q columns) transposed+converted from f32 x.
// Block tile 64(o) x 128(n), K=32 fully unrolled. grid = B*4*32 = 2048
// --------------------------------------------------------------------------
__global__ __launch_bounds__(256) void k_final(const __bf16* __restrict__ Mwh,
                                               const float* __restrict__ x,
                                               const float* __restrict__ cb,
                                               const float* __restrict__ gam,
                                               float* __restrict__ out) {
  __shared__ __bf16 sA[2][64 * LDST];
  __shared__ __bf16 sB[2][128 * LDST];   // [n][K]
  const int bx = blockIdx.x;
  const int b  = bx >> 7;              // 128 tiles per batch
  const int r0 = bx & 127;
  const int o0  = (r0 >> 5) * 64;
  const int nt0 = (r0 & 31) * 128;
  const __bf16* A = Mwh + (size_t)b * CDIM * CDIM;
  const float*  Q = x + (size_t)b * CDIM * NDIM;

  const int t = threadIdx.x, lane = t & 31, wid = t >> 5;
  const int m0 = (wid >> 2) * 32, n0 = (wid & 3) * 32;

  const int ar = t >> 2, akg = (t & 3) * 8;     // A: 64 rows x 32
  const __bf16* gA = A + (size_t)(o0 + ar) * CDIM + akg;
  const int la = ar * LDST + akg;
  const unsigned lAo[2] = {lds_off(&sA[0][la]), lds_off(&sA[1][la])};
  const int bk = t >> 3, bc = (t & 7) * 16;     // B: 32 K-rows x 128 n-cols

  auto stageB = [&](int buf, int kk) {
    const float* src = Q + (size_t)(kk + bk) * NDIM + nt0 + bc;
    f32x4 u0 = *(const f32x4*)(src);
    f32x4 u1 = *(const f32x4*)(src + 4);
    f32x4 u2 = *(const f32x4*)(src + 8);
    f32x4 u3 = *(const f32x4*)(src + 12);
#pragma unroll
    for (int i = 0; i < 4; ++i) {    // in-LDS transpose + f32->bf16 convert
      sB[buf][(bc + i) * LDST + bk]      = (__bf16)u0[i];
      sB[buf][(bc + 4 + i) * LDST + bk]  = (__bf16)u1[i];
      sB[buf][(bc + 8 + i) * LDST + bk]  = (__bf16)u2[i];
      sB[buf][(bc + 12 + i) * LDST + bk] = (__bf16)u3[i];
    }
  };

  async_b128(gA, lAo[0]);
  stageB(0, 0);
  wait_async0();
  __syncthreads();

  f32x8 acc00 = {}, acc01 = {}, acc10 = {}, acc11 = {};
#pragma unroll
  for (int kk = 0; kk < CDIM; kk += 32) {
    const int cur = (kk >> 5) & 1, nxt = cur ^ 1;
    const bool more = (kk + 32 < CDIM);
    if (more) {
      async_b128(gA + kk + 32, lAo[nxt]);
      if (kk + 64 < CDIM)
        __builtin_prefetch(Q + (size_t)(kk + 64 + bk) * NDIM + nt0 + bc, 0, 3);
    }
    bf16x16 fa0 = frag_a(&sA[cur][m0 * LDST], lane, LDST);
    bf16x16 fa1 = frag_a(&sA[cur][(m0 + 16) * LDST], lane, LDST);
    bf16x16 fb0 = frag_b(&sB[cur][n0 * LDST], lane, LDST);
    bf16x16 fb1 = frag_b(&sB[cur][(n0 + 16) * LDST], lane, LDST);
    acc00 = WMMA_BF16(fa0, fb0, acc00);
    acc01 = WMMA_BF16(fa0, fb1, acc01);
    acc10 = WMMA_BF16(fa1, fb0, acc10);
    acc11 = WMMA_BF16(fa1, fb1, acc11);
    if (more) stageB(nxt, kk + 32);
    wait_async0();
    __syncthreads();
  }

  const float g = gam[0];
  float* O = out + (size_t)b * CDIM * NDIM;
  const int half = lane >> 4, nl = lane & 15;
#pragma unroll
  for (int r = 0; r < 8; ++r) {
    const int mA = o0 + m0 + (half ? 8 : 0) + r;
    const int mB = mA + 16;
    const int col = nt0 + n0 + nl;
    const float bA = cb[mA], bB = cb[mB];
    const size_t iA = (size_t)mA * NDIM + col;
    const size_t iB = (size_t)mB * NDIM + col;
    O[iA]      = g * (acc00[r] + bA) + Q[iA];
    O[iA + 16] = g * (acc01[r] + bA) + Q[iA + 16];
    O[iB]      = g * (acc10[r] + bB) + Q[iB];
    O[iB + 16] = g * (acc11[r] + bB) + Q[iB + 16];
  }
}

extern "C" void kernel_launch(void* const* d_in, const int* in_sizes, int n_in,
                              void* d_out, int out_size, void* d_ws, size_t ws_size,
                              hipStream_t stream) {
  const float* x  = (const float*)d_in[0];
  const float* y  = (const float*)d_in[1];
  const float* z  = (const float*)d_in[2];
  const float* w  = (const float*)d_in[3];
  const float* cb = (const float*)d_in[4];
  const float* g  = (const float*)d_in[5];
  float* out = (float*)d_out;

  const size_t nel = (size_t)BDIM * CDIM * NDIM;       // 16,777,216
  __bf16* xh     = (__bf16*)d_ws;
  __bf16* yh     = xh + nel;
  __bf16* zh     = yh + nel;
  __bf16* wh     = zh + nel;                           // C*C
  __bf16* attnTh = wh + (size_t)CDIM * CDIM;           // B*C*C
  __bf16* Mwh    = attnTh + (size_t)BDIM * CDIM * CDIM;
  float*  E      = (float*)(Mwh + (size_t)BDIM * CDIM * CDIM);  // 3*B*C*C f32

  k_cvt    <<<2048,          256, 0, stream>>>(x, xh, (int)nel);
  k_cvt    <<<2048,          256, 0, stream>>>(y, yh, (int)nel);
  k_cvt    <<<2048,          256, 0, stream>>>(z, zh, (int)nel);
  k_cvt    <<<32,            256, 0, stream>>>(w, wh, CDIM * CDIM);
  k_energy <<<3 * BDIM * 8,  256, 0, stream>>>(xh, yh, zh, E);
  k_softmax<<<BDIM * CDIM,   256, 0, stream>>>(E, attnTh);
  k_mix    <<<BDIM * 8,      256, 0, stream>>>(wh, attnTh, Mwh);
  k_final  <<<BDIM * 4 * 32, 256, 0, stream>>>(Mwh, x, cb, g, out);
}
